// GroupIAMDecoder_83854941487558
// MI455X (gfx1250) — compile-verified, matrix operand order checked by reading, not run on hardware
//
#include <hip/hip_runtime.h>
#include <hip/hip_bf16.h>

typedef __bf16 bf16;
typedef __bf16 v16bf __attribute__((ext_vector_type(16)));
typedef float  v8f   __attribute__((ext_vector_type(8)));

#define HH 80
#define WWID 80
#define HWSZ 6400
#define BBATCH 8

__device__ __forceinline__ float bf2f(bf16 h) { return (float)h; }
__device__ __forceinline__ bf16  f2bf(float f) { return (bf16)f; }

// Wave-relative LDS byte offset of a __shared__ object: generic LDS pointers
// carry the LDS offset in their low 32 bits (aperture mapping, ISA 10.2);
// the async-to-LDS VDST VGPR expects exactly this (HW adds LDS_BASE).
__device__ __forceinline__ unsigned lds_offset(const void* p) {
    return (unsigned)(unsigned long long)p;
}

// Issue a per-lane async copy of 4 bytes from global memory into LDS.
__device__ __forceinline__ void async_copy_b32(unsigned lds_dst, const void* gsrc) {
    asm volatile("global_load_async_to_lds_b32 %0, %1, off"
                 :: "v"(lds_dst), "v"(gsrc) : "memory");
}
__device__ __forceinline__ void async_wait0() {
    asm volatile("s_wait_asynccnt 0x0" ::: "memory");
}

// Pair index inside an 8-VGPR 16-bit fragment row (CDNA5 WMMA layout):
// VGPR i holds K-pair at (i>=4?16:0) + (i&3)*2 + group*8, stored as u32 pairs.
__device__ __forceinline__ int frag_pair_idx(int i, int grp) {
    return ((i >= 4) ? 8 : 0) + (i & 3) + grp * 4;
}

union FragU { v16bf v; unsigned int u[8]; };

// ---------------------------------------------------------------------------
// Implicit-GEMM conv (3x3 pad=1 or 1x1 pad=0), optional groups.
// out[g*mg+m, p] = act( sum_k wgt[(g*mg+m)*K + k] * im2col(k, p) + bias )
// Block: 128 threads = 4 waves. Block tile: 256(M) x 64(N pixels), K-step 32.
// Wave w computes rows [w*64, w*64+64) as 4x4 grid of 16x16x32 bf16 WMMAs.
// Weight slab is fetched with GLOBAL_LOAD_ASYNC_TO_LDS_B32 when the tile is
// full (no zero-padding needed); scalar path covers tails / grouped conv.
// ---------------------------------------------------------------------------
__global__ __launch_bounds__(128)
void conv_wmma_kernel(const bf16* __restrict__ in, const bf16* __restrict__ wgt,
                      const float* __restrict__ bias,
                      bf16* __restrict__ outb, float* __restrict__ outf,
                      int cin_total, int cin_g, int mg, int cout_total,
                      int ks, int pad, int act)
{
    __shared__ __align__(16) unsigned int sA[256 * 16]; // [m_local][16 u32] = 32 bf16 K
    __shared__ __align__(16) bf16        sB[64 * 32];   // [n_local][k'] transposed im2col

    const int tid   = threadIdx.x;
    const int lane  = tid & 31;
    const int wave  = tid >> 5;
    const int grp   = lane >> 4;
    const int l15   = lane & 15;
    const int g     = blockIdx.z;
    const int ptile = blockIdx.x;            // 64-pixel tile over B*HW
    const int mblock = blockIdx.y * 256;
    const int kk2   = ks * ks;
    const int K_total = cin_g * kk2;
    const int KT    = (K_total + 31) >> 5;
    const unsigned sA_lds = lds_offset(sA);

    v8f acc[4][4];
    const v8f vzero = {0.f, 0.f, 0.f, 0.f, 0.f, 0.f, 0.f, 0.f};
    #pragma unroll
    for (int t = 0; t < 4; ++t)
        #pragma unroll
        for (int u = 0; u < 4; ++u) acc[t][u] = vzero;

    for (int kt = 0; kt < KT; ++kt) {
        const int k0 = kt << 5;
        const bool full = (mblock + 256 <= mg) && (k0 + 32 <= K_total);

        // ---- stage A (weights) : 256x32 bf16 ----
        if (full) {
            // async L2 -> LDS, no VGPR bounce (ASYNCcnt-tracked)
            #pragma unroll 4
            for (int it = 0; it < 32; ++it) {
                int idx = tid + (it << 7);           // u32 slot 0..4095
                int m_local = idx >> 4;
                int k = k0 + ((idx & 15) << 1);
                const bf16* src = wgt + (size_t)(g * mg + mblock + m_local) * K_total + k;
                async_copy_b32(sA_lds + (unsigned)idx * 4u, src);
            }
            async_wait0();
        } else {
            #pragma unroll 4
            for (int it = 0; it < 32; ++it) {
                int idx = tid + (it << 7);
                int m_local = idx >> 4;
                int k = k0 + ((idx & 15) << 1);
                int m = mblock + m_local;
                unsigned int val = 0u;
                if (m < mg && k < K_total) {
                    const bf16* src = wgt + (size_t)(g * mg + m) * K_total + k;
                    val = *(const unsigned int*)src;
                    __builtin_prefetch(src + 32, 0, 0);
                }
                sA[idx] = val;
            }
        }

        // ---- stage B (im2col activations) : 32x64 stored transposed [n][k'] ----
        #pragma unroll 4
        for (int it = 0; it < 16; ++it) {
            int e = tid + (it << 7);                 // 0..2047
            int n  = e >> 5;
            int kk = e & 31;
            int k  = k0 + kk;
            bf16 v = f2bf(0.f);
            if (k < K_total) {
                int ci = k / kk2;
                int r  = k - ci * kk2;
                int ky = r / ks;
                int kx = r - ky * ks;
                int p  = ptile * 64 + n;
                int b  = p / HWSZ;
                int hw = p - b * HWSZ;
                int y  = hw / WWID;
                int x  = hw - y * WWID;
                int iy = y + ky - pad;
                int ix = x + kx - pad;
                if (iy >= 0 && iy < HH && ix >= 0 && ix < WWID)
                    v = in[((size_t)b * cin_total + g * cin_g + ci) * HWSZ + iy * WWID + ix];
            }
            sB[n * 32 + kk] = v;
        }
        __syncthreads();

        // ---- fragments + 16 WMMAs per wave ----
        const unsigned int* sB32 = (const unsigned int*)sB;
        FragU afr[4], bfr[4];
        #pragma unroll
        for (int t = 0; t < 4; ++t) {
            const unsigned int* row = sA + (wave * 64 + t * 16 + l15) * 16;
            #pragma unroll
            for (int i = 0; i < 8; ++i) afr[t].u[i] = row[frag_pair_idx(i, grp)];
        }
        #pragma unroll
        for (int u = 0; u < 4; ++u) {
            const unsigned int* row = sB32 + (u * 16 + l15) * 16;
            #pragma unroll
            for (int i = 0; i < 8; ++i) bfr[u].u[i] = row[frag_pair_idx(i, grp)];
        }
        #pragma unroll
        for (int t = 0; t < 4; ++t)
            #pragma unroll
            for (int u = 0; u < 4; ++u)
                acc[t][u] = __builtin_amdgcn_wmma_f32_16x16x32_bf16(
                    false, afr[t].v, false, bfr[u].v, (short)0, acc[t][u], false, false);
        __syncthreads();
    }

    // ---- epilogue: bias + activation + store (C layout: VGPR r -> M=r+grp*8) ----
    #pragma unroll
    for (int t = 0; t < 4; ++t)
        #pragma unroll
        for (int u = 0; u < 4; ++u)
            #pragma unroll
            for (int r = 0; r < 8; ++r) {
                int m = mblock + wave * 64 + t * 16 + r + grp * 8;
                if (m < mg) {
                    int co = g * mg + m;
                    int p  = ptile * 64 + u * 16 + l15;
                    int b  = p / HWSZ;
                    int hw = p - b * HWSZ;
                    float v = acc[t][u][r] + bias[co];
                    if (act == 1)      v = v > 0.f ? v : 0.f;
                    else if (act == 2) v = 1.f / (1.f + __expf(-v));
                    size_t o = ((size_t)b * cout_total + co) * HWSZ + hw;
                    if (outb) outb[o] = f2bf(v);
                    else      outf[o] = v;
                }
            }
}

// ---------------------------------------------------------------------------
// Generic strided batched GEMM: C[z,m,n] = act( sum_k A[z,m,k] * B[z,n,k] + bias[n] )
// A contiguous in k (sak==1, async-to-LDS fast path). B has (sbn, sbk) strides.
// Block: 128 threads = 4 waves. Block tile: 128(M) x 64(N), K-step 32.
// Wave w computes rows [w*32, w*32+32) as 2x4 grid of 16x16x32 bf16 WMMAs.
// ---------------------------------------------------------------------------
__global__ __launch_bounds__(128)
void gemm_wmma_kernel(const bf16* __restrict__ A, size_t strideA, int sam,
                      const bf16* __restrict__ Bm, size_t strideB, int sbn, int sbk,
                      const float* __restrict__ bias,
                      float* __restrict__ Cf, bf16* __restrict__ Cb,
                      size_t strideC, int scm,
                      int M, int N, int K, int act)
{
    __shared__ __align__(16) bf16 sGA[128 * 32];
    __shared__ __align__(16) bf16 sGB[64 * 32];

    const int tid  = threadIdx.x;
    const int lane = tid & 31;
    const int wave = tid >> 5;
    const int grp  = lane >> 4;
    const int l15  = lane & 15;
    const int z    = blockIdx.z;
    const int nbase = blockIdx.x * 64;
    const int mbase = blockIdx.y * 128;
    const int KT = (K + 31) >> 5;
    const unsigned sGA_lds = lds_offset(sGA);

    v8f acc[2][4];
    const v8f vzero = {0.f, 0.f, 0.f, 0.f, 0.f, 0.f, 0.f, 0.f};
    #pragma unroll
    for (int t = 0; t < 2; ++t)
        #pragma unroll
        for (int u = 0; u < 4; ++u) acc[t][u] = vzero;

    for (int kt = 0; kt < KT; ++kt) {
        const int k0 = kt << 5;
        const bool full = (mbase + 128 <= M) && (k0 + 32 <= K);

        if (full) {
            // async L2 -> LDS for the A slab (2048 u32 slots)
            #pragma unroll 4
            for (int it = 0; it < 16; ++it) {
                int idx = tid + (it << 7);           // u32 slot 0..2047
                int ml = idx >> 4;
                int k  = k0 + ((idx & 15) << 1);
                const bf16* src = A + (size_t)z * strideA + (size_t)(mbase + ml) * sam + k;
                async_copy_b32(sGA_lds + (unsigned)idx * 4u, src);
            }
            async_wait0();
        } else {
            #pragma unroll 4
            for (int it = 0; it < 32; ++it) {        // A: 128x32 = 4096 elems
                int e = tid + (it << 7);
                int ml = e >> 5, kk = e & 31;
                int m = mbase + ml, k = k0 + kk;
                bf16 v = f2bf(0.f);
                if (m < M && k < K) v = A[(size_t)z * strideA + (size_t)m * sam + k];
                sGA[ml * 32 + kk] = v;
            }
        }
        #pragma unroll 4
        for (int it = 0; it < 16; ++it) {            // B: 64x32 = 2048 elems
            int e = tid + (it << 7);
            int nl = e >> 5, kk = e & 31;
            int n = nbase + nl, k = k0 + kk;
            bf16 v = f2bf(0.f);
            if (n < N && k < K) v = Bm[(size_t)z * strideB + (size_t)n * sbn + (size_t)k * sbk];
            sGB[nl * 32 + kk] = v;
        }
        __syncthreads();

        const unsigned int* a32 = (const unsigned int*)sGA;
        const unsigned int* b32 = (const unsigned int*)sGB;
        FragU afr[2], bfr[4];
        #pragma unroll
        for (int t = 0; t < 2; ++t) {
            const unsigned int* row = a32 + (wave * 32 + t * 16 + l15) * 16;
            #pragma unroll
            for (int i = 0; i < 8; ++i) afr[t].u[i] = row[frag_pair_idx(i, grp)];
        }
        #pragma unroll
        for (int u = 0; u < 4; ++u) {
            const unsigned int* row = b32 + (u * 16 + l15) * 16;
            #pragma unroll
            for (int i = 0; i < 8; ++i) bfr[u].u[i] = row[frag_pair_idx(i, grp)];
        }
        #pragma unroll
        for (int t = 0; t < 2; ++t)
            #pragma unroll
            for (int u = 0; u < 4; ++u)
                acc[t][u] = __builtin_amdgcn_wmma_f32_16x16x32_bf16(
                    false, afr[t].v, false, bfr[u].v, (short)0, acc[t][u], false, false);
        __syncthreads();
    }

    #pragma unroll
    for (int t = 0; t < 2; ++t)
        #pragma unroll
        for (int u = 0; u < 4; ++u)
            #pragma unroll
            for (int r = 0; r < 8; ++r) {
                int m = mbase + wave * 32 + t * 16 + r + grp * 8;
                int n = nbase + u * 16 + l15;
                if (m < M && n < N) {
                    float v = acc[t][u][r] + (bias ? bias[n] : 0.f);
                    if (act == 1) v = v > 0.f ? v : 0.f;
                    size_t o = (size_t)z * strideC + (size_t)m * scm + n;
                    if (Cf) Cf[o] = v;
                    else    Cb[o] = f2bf(v);
                }
            }
}

// ---------------------------------------------------------------------------
// Small glue kernels
// ---------------------------------------------------------------------------
__global__ void f32_to_bf16_kernel(const float* __restrict__ s, bf16* __restrict__ d, int n) {
    int i = blockIdx.x * blockDim.x + threadIdx.x;
    if (i < n) d[i] = f2bf(s[i]);
}

// f[b, 0..1, hw] = coords, f[b, 2+c, hw] = features[b, c, hw]; bf16 out.
__global__ void build_f_kernel(const float* __restrict__ feat, bf16* __restrict__ f) {
    int i = blockIdx.x * blockDim.x + threadIdx.x;
    const int CT = 258;
    int total = BBATCH * CT * HWSZ;
    if (i >= total) return;
    int b  = i / (CT * HWSZ);
    int r  = i - b * CT * HWSZ;
    int c  = r / HWSZ;
    int hw = r - c * HWSZ;
    float v;
    if (c == 0)      { int x = hw % WWID; v = -1.f + 2.f * (float)x / (float)(WWID - 1); }
    else if (c == 1) { int y = hw / WWID; v = -1.f + 2.f * (float)y / (float)(HH - 1); }
    else             v = feat[((size_t)b * 256 + (c - 2)) * HWSZ + hw];
    f[i] = f2bf(v);
}

// One block per IAM row: sum of sigmoid activations (prob already holds sigmoid).
__global__ void rowsum_kernel(const bf16* __restrict__ prob, float* __restrict__ rs) {
    __shared__ float red[256];
    int row = blockIdx.x;
    const bf16* p = prob + (size_t)row * HWSZ;
    float s = 0.f;
    for (int l = threadIdx.x; l < HWSZ; l += 256) s += bf2f(p[l]);
    red[threadIdx.x] = s;
    __syncthreads();
    for (int st = 128; st > 0; st >>= 1) {
        if (threadIdx.x < st) red[threadIdx.x] += red[threadIdx.x + st];
        __syncthreads();
    }
    if (threadIdx.x == 0) rs[row] = red[0];
}

// pooled[b, g*100+n, c] / max(rowsum, 1e-6) -> inst[b, n, g*256+c] (bf16)
__global__ void regroup_kernel(const float* __restrict__ pooled,
                               const float* __restrict__ rs,
                               bf16* __restrict__ out) {
    int i = blockIdx.x * blockDim.x + threadIdx.x;
    if (i >= BBATCH * 100 * 1024) return;
    int b   = i / 102400;
    int r   = i - b * 102400;
    int n   = r / 1024;
    int cc  = r - n * 1024;
    int g   = cc >> 8;
    int c   = cc & 255;
    int row = b * 400 + g * 100 + n;
    float v = pooled[(size_t)b * 102400 + (size_t)(g * 100 + n) * 256 + c]
              / fmaxf(rs[row], 1e-6f);
    out[i] = f2bf(v);
}

// Bilinear 2x upsample, half-pixel centers: [B*100,80,80] -> [B*100,160,160]
__global__ void upsample_kernel(const float* __restrict__ lo, float* __restrict__ out) {
    int i = blockIdx.x * blockDim.x + threadIdx.x;
    const int OH = 160, OW = 160;
    int total = BBATCH * 100 * OH * OW;
    if (i >= total) return;
    int img = i / (OH * OW);
    int r   = i - img * (OH * OW);
    int oy  = r / OW;
    int ox  = r - oy * OW;
    float fy = (float)oy * 0.5f - 0.25f;
    float fx = (float)ox * 0.5f - 0.25f;
    int y0 = (int)floorf(fy), x0 = (int)floorf(fx);
    float ty = fy - (float)y0, tx = fx - (float)x0;
    int y0c = min(max(y0, 0), HH - 1),  y1c = min(max(y0 + 1, 0), HH - 1);
    int x0c = min(max(x0, 0), WWID - 1), x1c = min(max(x0 + 1, 0), WWID - 1);
    const float* p = lo + (size_t)img * HWSZ;
    float v00 = p[y0c * WWID + x0c], v01 = p[y0c * WWID + x1c];
    float v10 = p[y1c * WWID + x0c], v11 = p[y1c * WWID + x1c];
    out[i] = (1.f - ty) * ((1.f - tx) * v00 + tx * v01)
           +        ty  * ((1.f - tx) * v10 + tx * v11);
}

// ---------------------------------------------------------------------------
extern "C" void kernel_launch(void* const* d_in, const int* in_sizes, int n_in,
                              void* d_out, int out_size, void* d_ws, size_t ws_size,
                              hipStream_t stream) {
    (void)in_sizes; (void)n_in; (void)out_size; (void)ws_size;
    const float* features = (const float*)d_in[0];
    const float* inst_w0  = (const float*)d_in[1];
    const float* inst_b0  = (const float*)d_in[2];
    const float* inst_ws  = (const float*)d_in[3];
    const float* inst_bs  = (const float*)d_in[4];
    const float* iam_w    = (const float*)d_in[5];
    const float* iam_b    = (const float*)d_in[6];
    const float* fc_w     = (const float*)d_in[7];
    const float* fc_b     = (const float*)d_in[8];
    const float* cls_w    = (const float*)d_in[9];
    const float* cls_b    = (const float*)d_in[10];
    const float* ker_w    = (const float*)d_in[11];
    const float* ker_b    = (const float*)d_in[12];
    const float* obj_w    = (const float*)d_in[13];
    const float* obj_b    = (const float*)d_in[14];
    const float* mask_w0  = (const float*)d_in[15];
    const float* mask_b0  = (const float*)d_in[16];
    const float* mask_ws  = (const float*)d_in[17];
    const float* mask_bs  = (const float*)d_in[18];
    const float* proj_w   = (const float*)d_in[19];
    const float* proj_b   = (const float*)d_in[20];

    float* out_logits = (float*)d_out;                               // [800, 80]
    float* out_masks  = out_logits + 8 * 100 * 80;                   // [800, 160,160]
    float* out_scores = out_masks + (size_t)8 * 100 * 160 * 160;     // [800, 1]

    char* ws = (char*)d_ws;
    size_t off = 0;
    auto alloc = [&](size_t bytes) -> char* {
        off = (off + 255) & ~(size_t)255;
        char* p = ws + off;
        off += bytes;
        return p;
    };

    bf16* f_bf   = (bf16*)alloc((size_t)8 * 258 * HWSZ * 2);
    bf16* actA   = (bf16*)alloc((size_t)8 * 256 * HWSZ * 2);
    bf16* actB   = (bf16*)alloc((size_t)8 * 256 * HWSZ * 2);
    bf16* mA     = (bf16*)alloc((size_t)8 * 256 * HWSZ * 2);
    bf16* mB     = (bf16*)alloc((size_t)8 * 256 * HWSZ * 2);
    bf16* prob   = (bf16*)alloc((size_t)8 * 400 * HWSZ * 2);
    float* rsum  = (float*)alloc((size_t)3200 * 4);
    float* pooled= (float*)alloc((size_t)8 * 400 * 256 * 4);
    bf16* instrg = (bf16*)alloc((size_t)800 * 1024 * 2);
    bf16* fc_out = (bf16*)alloc((size_t)800 * 1024 * 2);
    bf16* ker_out= (bf16*)alloc((size_t)800 * 128 * 2);
    bf16* mfeat  = (bf16*)alloc((size_t)8 * 128 * HWSZ * 2);
    float* lowres= (float*)alloc((size_t)8 * 100 * HWSZ * 4);

    bf16* w_i0  = (bf16*)alloc((size_t)256 * 2322 * 2);
    bf16* w_is  = (bf16*)alloc((size_t)3 * 256 * 2304 * 2);
    bf16* w_iam = (bf16*)alloc((size_t)400 * 576 * 2);
    bf16* w_fc  = (bf16*)alloc((size_t)1024 * 1024 * 2);
    bf16* w_cls = (bf16*)alloc((size_t)80 * 1024 * 2);
    bf16* w_ker = (bf16*)alloc((size_t)128 * 1024 * 2);
    bf16* w_obj = (bf16*)alloc((size_t)1 * 1024 * 2);
    bf16* w_m0  = (bf16*)alloc((size_t)256 * 2322 * 2);
    bf16* w_ms  = (bf16*)alloc((size_t)3 * 256 * 2304 * 2);
    bf16* w_pr  = (bf16*)alloc((size_t)128 * 256 * 2);

    auto cvt = [&](const float* s, bf16* d, int n) {
        f32_to_bf16_kernel<<<(n + 255) / 256, 256, 0, stream>>>(s, d, n);
    };
    cvt(inst_w0, w_i0, 256 * 2322);
    cvt(inst_ws, w_is, 3 * 256 * 2304);
    cvt(iam_w,   w_iam, 400 * 576);
    cvt(fc_w,    w_fc, 1024 * 1024);
    cvt(cls_w,   w_cls, 80 * 1024);
    cvt(ker_w,   w_ker, 128 * 1024);
    cvt(obj_w,   w_obj, 1024);
    cvt(mask_w0, w_m0, 256 * 2322);
    cvt(mask_ws, w_ms, 3 * 256 * 2304);
    cvt(proj_w,  w_pr, 128 * 256);

    {
        int n = 8 * 258 * HWSZ;
        build_f_kernel<<<(n + 255) / 256, 256, 0, stream>>>(features, f_bf);
    }

    const dim3 blk(128);
    auto conv = [&](const bf16* in, const bf16* w, const float* b,
                    bf16* ob, float* of, int cint, int cing, int mg, int coutt,
                    int ks, int pad, int act, int groups) {
        dim3 g(8 * HWSZ / 64, (unsigned)((mg + 255) / 256), (unsigned)groups);
        conv_wmma_kernel<<<g, blk, 0, stream>>>(in, w, b, ob, of,
                                                cint, cing, mg, coutt, ks, pad, act);
    };

    // ---- instance branch conv stack ----
    conv(f_bf, w_i0, inst_b0, actA, nullptr, 258, 258, 256, 256, 3, 1, 1, 1);
    conv(actA, w_is + (size_t)0 * 256 * 2304, inst_bs + 0 * 256, actB, nullptr, 256, 256, 256, 256, 3, 1, 1, 1);
    conv(actB, w_is + (size_t)1 * 256 * 2304, inst_bs + 1 * 256, actA, nullptr, 256, 256, 256, 256, 3, 1, 1, 1);
    conv(actA, w_is + (size_t)2 * 256 * 2304, inst_bs + 2 * 256, actB, nullptr, 256, 256, 256, 256, 3, 1, 1, 1);
    // z = actB [8,256,6400]

    // ---- grouped IAM conv, sigmoid epilogue -> prob bf16 ----
    conv(actB, w_iam, iam_b, prob, nullptr, 256, 64, 100, 400, 3, 1, 2, 4);
    rowsum_kernel<<<3200, 256, 0, stream>>>(prob, rsum);

    // ---- pooling einsum: pooled[b, 400, 256] = prob[b] @ z[b]^T ----
    {
        dim3 g((256 + 63) / 64, (400 + 127) / 128, 8);
        gemm_wmma_kernel<<<g, blk, 0, stream>>>(
            prob, (size_t)400 * HWSZ, HWSZ,
            actB, (size_t)256 * HWSZ, HWSZ, 1,
            nullptr, pooled, nullptr, (size_t)400 * 256, 256,
            400, 256, HWSZ, 0);
    }
    {
        int n = 800 * 1024;
        regroup_kernel<<<(n + 255) / 256, 256, 0, stream>>>(pooled, rsum, instrg);
    }

    // ---- FC + heads ----
    {
        dim3 g(1024 / 64, (800 + 127) / 128, 1);
        gemm_wmma_kernel<<<g, blk, 0, stream>>>(
            instrg, 0, 1024, w_fc, 0, 1024, 1, fc_b,
            nullptr, fc_out, 0, 1024, 800, 1024, 1024, 1);
    }
    {
        dim3 g((80 + 63) / 64, (800 + 127) / 128, 1);
        gemm_wmma_kernel<<<g, blk, 0, stream>>>(
            fc_out, 0, 1024, w_cls, 0, 1024, 1, cls_b,
            out_logits, nullptr, 0, 80, 800, 80, 1024, 0);
    }
    {
        dim3 g((128 + 63) / 64, (800 + 127) / 128, 1);
        gemm_wmma_kernel<<<g, blk, 0, stream>>>(
            fc_out, 0, 1024, w_ker, 0, 1024, 1, ker_b,
            nullptr, ker_out, 0, 128, 800, 128, 1024, 0);
    }
    {
        dim3 g(1, (800 + 127) / 128, 1);
        gemm_wmma_kernel<<<g, blk, 0, stream>>>(
            fc_out, 0, 1024, w_obj, 0, 1024, 1, obj_b,
            out_scores, nullptr, 0, 1, 800, 1, 1024, 0);
    }

    // ---- mask branch ----
    conv(f_bf, w_m0, mask_b0, mA, nullptr, 258, 258, 256, 256, 3, 1, 1, 1);
    conv(mA, w_ms + (size_t)0 * 256 * 2304, mask_bs + 0 * 256, mB, nullptr, 256, 256, 256, 256, 3, 1, 1, 1);
    conv(mB, w_ms + (size_t)1 * 256 * 2304, mask_bs + 1 * 256, mA, nullptr, 256, 256, 256, 256, 3, 1, 1, 1);
    conv(mA, w_ms + (size_t)2 * 256 * 2304, mask_bs + 2 * 256, mB, nullptr, 256, 256, 256, 256, 3, 1, 1, 1);
    conv(mB, w_pr, proj_b, mfeat, nullptr, 256, 256, 128, 128, 1, 0, 0, 1);

    // ---- pred_masks = pred_kernel @ mask_feat, per batch ----
    {
        dim3 g(HWSZ / 64, 1, 8);
        gemm_wmma_kernel<<<g, blk, 0, stream>>>(
            ker_out, (size_t)100 * 128, 128,
            mfeat, (size_t)128 * HWSZ, 1, HWSZ,
            nullptr, lowres, nullptr, (size_t)100 * HWSZ, HWSZ,
            100, HWSZ, 128, 0);
    }
    {
        int n = 8 * 100 * 160 * 160;
        upsample_kernel<<<(n + 255) / 256, 256, 0, stream>>>(lowres, out_masks);
    }
}
